// Decoder_14448269984012
// MI455X (gfx1250) — matrix operand order repaired and checked by reading, not compiled, not run on hardware
//
#include <hip/hip_runtime.h>
#include <cstdint>
#include <cstddef>

typedef __attribute__((ext_vector_type(16))) __bf16 bf16x16;
typedef __attribute__((ext_vector_type(8)))  __bf16 bf16x8;
typedef __attribute__((ext_vector_type(4)))  __bf16 bf16x4;
typedef __attribute__((ext_vector_type(8)))  float  f32x8;

__device__ __forceinline__ float sigmf(float x) { return 1.0f / (1.0f + __expf(-x)); }

__device__ __forceinline__ bf16x16 make_frag(const __bf16* c0, const __bf16* c1) {
  bf16x8 lo = *(const bf16x8*)c0;
  bf16x8 hi = *(const bf16x8*)c1;
  bf16x16 r;
#pragma unroll
  for (int i = 0; i < 8; ++i) { r[i] = lo[i]; r[8 + i] = hi[i]; }
  return r;
}

__device__ __forceinline__ f32x8 wmma_bf16(bf16x16 a, bf16x16 b, f32x8 c) {
  return __builtin_amdgcn_wmma_f32_16x16x32_bf16(false, a, false, b, (short)0, c, false, false);
}

// ---------------------------------------------------------------------------
// Generic D = A * W^T (+bias)(+relu) GEMM.
// fp32 inputs converted to bf16 during LDS staging; WMMA bf16, fp32 accumulate.
// Register double-buffer: next K-chunk's global loads issued while current
// chunk's ds_load + v_wmma execute.
// A may be split along K: cols [0,ksplit) from Aa, [ksplit,K) from Ab.
// Block tile 64x64, BK=32, 128 threads = 4 waves, each wave 32x32.
// Requires: M%64==0 (grid.y), N%64==0, K%32==0, ksplit%32==0.
// ---------------------------------------------------------------------------
template <bool SPLIT, bool HAS_BIAS, bool RELU>
__global__ __launch_bounds__(128) void gemm_bf16_wmma(
    const float* __restrict__ Aa, int ldAa,
    const float* __restrict__ Ab, int ldAb, int ksplit,
    const float* __restrict__ W, int ldW,
    const float* __restrict__ bias,
    float* __restrict__ D, int N, int K) {
  __shared__ __bf16 As[64][40];  // row stride 40 (80B): 16B-aligned chunks
  __shared__ __bf16 Bs[64][40];
  const int tid  = threadIdx.x;
  const int lane = tid & 31, wv = tid >> 5;
  const int m_off = (wv & 1) * 32, n_off = (wv >> 1) * 32;
  const int m0 = blockIdx.y * 64, n0 = blockIdx.x * 64;
  const int l15 = lane & 15, hb = lane >> 4;

  f32x8 acc00 = {}, acc01 = {}, acc10 = {}, acc11 = {};
  float4 ra[4], rw[4];

  auto load_regs = [&](int kc) {
    const float* Asrc = Aa; int ldA = ldAa, cbase = kc;
    if (SPLIT && kc >= ksplit) { Asrc = Ab; ldA = ldAb; cbase = kc - ksplit; }
#pragma unroll
    for (int it = 0; it < 4; ++it) {
      int idx = tid + it * 128;          // 512 = 64 rows x 8 float4
      int r = idx >> 3, c4 = idx & 7;
      ra[it] = *(const float4*)(Asrc + (size_t)(m0 + r) * ldA + cbase + c4 * 4);
      rw[it] = *(const float4*)(W + (size_t)(n0 + r) * ldW + kc + c4 * 4);
    }
  };
  auto store_lds = [&]() {
#pragma unroll
    for (int it = 0; it < 4; ++it) {
      int idx = tid + it * 128;
      int r = idx >> 3, c4 = idx & 7;
      bf16x4 pa; pa[0] = (__bf16)ra[it].x; pa[1] = (__bf16)ra[it].y;
                 pa[2] = (__bf16)ra[it].z; pa[3] = (__bf16)ra[it].w;
      *(bf16x4*)&As[r][c4 * 4] = pa;
      bf16x4 pw; pw[0] = (__bf16)rw[it].x; pw[1] = (__bf16)rw[it].y;
                 pw[2] = (__bf16)rw[it].z; pw[3] = (__bf16)rw[it].w;
      *(bf16x4*)&Bs[r][c4 * 4] = pw;
    }
  };

  load_regs(0);
  for (int kc = 0; kc < K; kc += 32) {
    store_lds();
    __syncthreads();
    if (kc + 32 < K) load_regs(kc + 32);   // in flight during ds_load + wmma
    const int koff = hb * 8;
    bf16x16 a0 = make_frag(&As[m_off + l15][hb * 8],      &As[m_off + l15][16 + hb * 8]);
    bf16x16 a1 = make_frag(&As[m_off + 16 + l15][hb * 8], &As[m_off + 16 + l15][16 + hb * 8]);
    bf16x16 b0 = make_frag(&Bs[n_off + l15][koff],        &Bs[n_off + l15][koff + 16]);
    bf16x16 b1 = make_frag(&Bs[n_off + 16 + l15][koff],   &Bs[n_off + 16 + l15][koff + 16]);
    acc00 = wmma_bf16(a0, b0, acc00);
    acc01 = wmma_bf16(a0, b1, acc01);
    acc10 = wmma_bf16(a1, b0, acc10);
    acc11 = wmma_bf16(a1, b1, acc11);
    __syncthreads();
  }

  auto emit = [&](const f32x8& a, int mi, int nj) {
    const int n = n0 + n_off + nj * 16 + l15;
    const float bv = HAS_BIAS ? bias[n] : 0.0f;
#pragma unroll
    for (int r = 0; r < 8; ++r) {
      int m = m0 + m_off + mi * 16 + hb * 8 + r;
      float v = a[r] + bv;
      if (RELU) v = v > 0.0f ? v : 0.0f;
      D[(size_t)m * N + n] = v;
    }
  };
  emit(acc00, 0, 0); emit(acc01, 0, 1); emit(acc10, 1, 0); emit(acc11, 1, 1);
}

// ---------------------------------------------------------------------------
// Fused LSTM step: gates = pre + A[32,K] * W[4096,K]^T + bias, then pointwise.
// 64 blocks (16 hidden units each), 128 threads; wave g owns gate g.
// Gate exchange via LDS, then c/h update, h written fp32 + bf16 (1-2 dests).
// Fragments loaded straight from global bf16 (weights pre-converted; every
// element used once per block, so no LDS staging for tiles).
// ---------------------------------------------------------------------------
template <bool HAS_PRE, bool HAS_BIAS, bool HAS_OUTB>
__global__ __launch_bounds__(128) void lstm_step(
    const __bf16* __restrict__ A, int K,
    const __bf16* __restrict__ W,
    const float* __restrict__ pre,   // [32,4096] (HAS_PRE)
    const float* __restrict__ bias,  // [4096]    (HAS_BIAS)
    float* __restrict__ Cst,         // [32,1024] in/out
    float* __restrict__ Hf32,        // [32,1024]
    __bf16* __restrict__ outA, int ldOutA,
    __bf16* __restrict__ outB, int ldOutB) {
  __shared__ float Gs[4][32][16];
  const int tid = threadIdx.x, lane = tid & 31, g = tid >> 5;
  const int n0 = blockIdx.x * 16;
  const int l15 = lane & 15, hb = lane >> 4;

  f32x8 acc0 = {}, acc1 = {};
  const __bf16* Wrow = W + (size_t)(g * 1024 + n0 + l15) * K + hb * 8;
  const __bf16* Ar0  = A + (size_t)l15 * K + hb * 8;
  const __bf16* Ar1  = A + (size_t)(16 + l15) * K + hb * 8;
  for (int kc = 0; kc < K; kc += 32) {
    bf16x16 a0 = make_frag(Ar0 + kc, Ar0 + kc + 16);
    bf16x16 a1 = make_frag(Ar1 + kc, Ar1 + kc + 16);
    bf16x16 b  = make_frag(Wrow + kc, Wrow + kc + 16);
    acc0 = wmma_bf16(a0, b, acc0);
    acc1 = wmma_bf16(a1, b, acc1);
  }

  const int col = g * 1024 + n0 + l15;
  const float badd = HAS_BIAS ? bias[col] : 0.0f;
#pragma unroll
  for (int r = 0; r < 8; ++r) {
    int mr = hb * 8 + r;
    float v0 = acc0[r] + badd + (HAS_PRE ? pre[(size_t)mr * 4096 + col] : 0.0f);
    float v1 = acc1[r] + badd + (HAS_PRE ? pre[(size_t)(mr + 16) * 4096 + col] : 0.0f);
    Gs[g][mr][l15]      = v0;
    Gs[g][mr + 16][l15] = v1;
  }
  __syncthreads();
  for (int idx = tid; idx < 512; idx += 128) {
    int m = idx >> 4, n = idx & 15;
    float gi = Gs[0][m][n], gf = Gs[1][m][n], gg = Gs[2][m][n], go = Gs[3][m][n];
    int ci = m * 1024 + n0 + n;
    float c = sigmf(gf) * Cst[ci] + sigmf(gi) * tanhf(gg);
    Cst[ci] = c;
    float h = sigmf(go) * tanhf(c);
    Hf32[ci] = h;
    outA[m * ldOutA + n0 + n] = (__bf16)h;
    if (HAS_OUTB) outB[m * ldOutB + n0 + n] = (__bf16)h;
  }
}

// ---------------------------------------------------------------------------
// Attention: p / win_start per (t,b) row.
// ---------------------------------------------------------------------------
__global__ __launch_bounds__(128) void k_pred(
    const float* __restrict__ attnp,  // [2048,512], already + attn_b1
    const float* __restrict__ aw2, const float* __restrict__ ab2,
    const int* __restrict__ lengths,
    float* __restrict__ p_arr, int* __restrict__ ws_arr) {
  __shared__ float red[128];
  int m = blockIdx.x, tid = threadIdx.x;
  float s = 0.0f;
  for (int j = tid; j < 512; j += 128)
    s += tanhf(attnp[(size_t)m * 512 + j]) * aw2[j];
  red[tid] = s;
  __syncthreads();
  for (int off = 64; off > 0; off >>= 1) {
    if (tid < off) red[tid] += red[tid + off];
    __syncthreads();
  }
  if (tid == 0) {
    float z  = red[0] + ab2[0];
    float sg = 1.0f / (1.0f + __expf(-z));
    int   b  = m & 31;
    float p  = 50.0f + (float)lengths[b] * sg;
    p_arr[m]  = p;
    ws_arr[m] = (int)rintf(p - 50.0f);
  }
}

// ---------------------------------------------------------------------------
// Attention: score over last-step window, mask->EPS, softmax*gaussian, context.
// ---------------------------------------------------------------------------
__global__ __launch_bounds__(128) void k_attn(
    const float* __restrict__ enc,   // [229,32,1024]
    const float* __restrict__ y1,    // [2048,1024]
    const float* __restrict__ p_arr, const int* __restrict__ ws_arr,
    const int* __restrict__ lengths,
    float* __restrict__ ctx) {       // [2048,1024]
  __shared__ float sc[101];
  __shared__ float av[101];
  __shared__ float mx_s, sum_s;
  int m = blockIdx.x, tid = threadIdx.x;
  int lane = tid & 31, wv = tid >> 5;
  int b   = m & 31;
  int wst = ws_arr[m];
  int wsl = ws_arr[2016 + b];        // (T-1)*B + b
  float p = p_arr[m];
  int len = lengths[b];

  const float* yr = y1 + (size_t)m * 1024;
  for (int w = wv; w < 101; w += 4) {
    const float* er = enc + ((size_t)(wsl + w) * 32 + b) * 1024;
    float acc = 0.0f;
    for (int h = lane; h < 1024; h += 32) acc += yr[h] * er[h];
    for (int off = 16; off > 0; off >>= 1) acc += __shfl_xor(acc, off, 32);
    if (lane == 0) sc[w] = acc;
  }
  __syncthreads();
  if (tid < 101) {
    int w = tid;
    if (w < (50 - wst) || w >= (len + 50 - wst)) sc[w] = 1e-14f;
  }
  __syncthreads();
  if (tid == 0) {
    float mx = -1e30f;
    for (int w = 0; w < 101; ++w) mx = fmaxf(mx, sc[w]);
    float sm = 0.0f;
    for (int w = 0; w < 101; ++w) sm += __expf(sc[w] - mx);
    mx_s = mx; sum_s = sm;
  }
  __syncthreads();
  if (tid < 101) {
    int w = tid;
    float d = (float)(wst + w) - p;
    float gauss = __expf(-d * d * (1.0f / 1250.0f));   // 2*STD2 = 1250
    av[w] = __expf(sc[w] - mx_s) / sum_s * gauss;
  }
  __syncthreads();
  for (int h = tid; h < 1024; h += 128) {
    float acc = 0.0f;
    for (int w = 0; w < 101; ++w)
      acc += av[w] * enc[((size_t)(wsl + w) * 32 + b) * 1024 + h];
    ctx[(size_t)m * 1024 + h] = acc;
  }
}

// ----------------------------- small helpers -------------------------------
__global__ void k_bias2(const float* a0, const float* b0, float* o0,
                        const float* a1, const float* b1, float* o1, int n) {
  int i = blockIdx.x * blockDim.x + threadIdx.x;
  if (i < n) { o0[i] = a0[i] + b0[i]; o1[i] = a1[i] + b1[i]; }
}

__global__ void k_cvt(const float* __restrict__ src, int lds,
                      __bf16* __restrict__ dst, int ldd, int rows, int cols) {
  int i = blockIdx.x * blockDim.x + threadIdx.x;
  if (i < rows * cols) {
    int r = i / cols, c = i % cols;
    dst[(size_t)r * ldd + c] = (__bf16)src[(size_t)r * lds + c];
  }
}

__global__ void k_emb(const float* __restrict__ emb, const int* __restrict__ tw,
                      float* __restrict__ out) {
  int i = blockIdx.x * blockDim.x + threadIdx.x;   // < 2048*1024
  int m = i >> 10, h = i & 1023;
  out[i] = emb[(size_t)tw[m] * 1024 + h];
}

__global__ void k_init(const float* __restrict__ h0, const float* __restrict__ c0,
                       float* h0s, float* c0s, float* c1s,
                       __bf16* hb0, __bf16* ac0) {
  int i = blockIdx.x * blockDim.x + threadIdx.x;   // < 32768
  int b = i >> 10, n = i & 1023;
  h0s[i] = h0[i];
  c0s[i] = c0[i];
  c1s[i] = c0[32768 + i];
  hb0[i] = (__bf16)h0[i];
  ac0[b * 2048 + 1024 + n] = (__bf16)h0[32768 + i];
}

__global__ void k_fin(const float* h0s, const float* y1last,
                      const float* c0s, const float* c1s,
                      float* hn, float* cn) {
  int i = blockIdx.x * blockDim.x + threadIdx.x;   // < 32768
  hn[i]         = h0s[i];
  hn[32768 + i] = y1last[i];
  cn[i]         = c0s[i];
  cn[32768 + i] = c1s[i];
}

// ---------------------------------------------------------------------------
extern "C" void kernel_launch(void* const* d_in, const int* in_sizes, int n_in,
                              void* d_out, int out_size, void* d_ws, size_t ws_size,
                              hipStream_t stream) {
  const float* enc    = (const float*)d_in[0];
  const float* ctx_in = (const float*)d_in[1];
  const float* h0     = (const float*)d_in[2];
  const float* c0     = (const float*)d_in[3];
  const float* emb    = (const float*)d_in[4];
  const float* aw1    = (const float*)d_in[5];
  const float* ab1    = (const float*)d_in[6];
  const float* aw2    = (const float*)d_in[7];
  const float* ab2    = (const float*)d_in[8];
  const float* Wih0   = (const float*)d_in[9];
  const float* Whh0   = (const float*)d_in[10];
  const float* bih0   = (const float*)d_in[11];
  const float* bhh0   = (const float*)d_in[12];
  const float* Wih1   = (const float*)d_in[13];
  const float* Whh1   = (const float*)d_in[14];
  const float* bih1   = (const float*)d_in[15];
  const float* bhh1   = (const float*)d_in[16];
  const float* fc1w   = (const float*)d_in[17];
  const float* fc1b   = (const float*)d_in[18];
  const float* fc2w   = (const float*)d_in[19];
  const float* fc2b   = (const float*)d_in[20];
  const int*   tw     = (const int*)d_in[21];
  const int*   lens   = (const int*)d_in[22];
  float* dout = (float*)d_out;

  char* wp = (char*)d_ws;
  auto carve = [&](size_t bytes) -> char* {
    char* r = wp; wp += (bytes + 255) & ~(size_t)255; return r;
  };
  float*  Xproj  = (float*)carve((size_t)2048 * 4096 * 4);
  float*  EmbX   = (float*)carve((size_t)2048 * 1024 * 4);
  float*  y1     = (float*)carve((size_t)2048 * 1024 * 4);
  float*  attnp  = (float*)carve((size_t)2048 * 512 * 4);
  float*  fc1o   = (float*)carve((size_t)2048 * 1024 * 4);
  __bf16* Whh0bf = (__bf16*)carve((size_t)4096 * 1024 * 2);
  __bf16* W1cat  = (__bf16*)carve((size_t)4096 * 2048 * 2);
  __bf16* hb0    = (__bf16*)carve((size_t)32 * 1024 * 2);
  __bf16* hb1    = (__bf16*)carve((size_t)32 * 1024 * 2);
  __bf16* ac0    = (__bf16*)carve((size_t)32 * 2048 * 2);
  __bf16* ac1    = (__bf16*)carve((size_t)32 * 2048 * 2);
  float*  h0s    = (float*)carve((size_t)32 * 1024 * 4);
  float*  c0s    = (float*)carve((size_t)32 * 1024 * 4);
  float*  c1s    = (float*)carve((size_t)32 * 1024 * 4);
  float*  b0c    = (float*)carve(4096 * 4);
  float*  b1c    = (float*)carve(4096 * 4);
  float*  p_arr  = (float*)carve(2048 * 4);
  int*    ws_arr = (int*)carve(2048 * 4);
  __bf16* hb[2] = {hb0, hb1};
  __bf16* ac[2] = {ac0, ac1};

  const size_t OFF_HN  = (size_t)2048 * 32000;
  const size_t OFF_CN  = OFF_HN + 65536;
  const size_t OFF_CTX = OFF_CN + 65536;
  float* ctx_out = dout + OFF_CTX;

  // Prep: combined biases, initial states, embedding gather, bf16 LSTM weights
  k_bias2<<<32, 128, 0, stream>>>(bih0, bhh0, b0c, bih1, bhh1, b1c, 4096);
  k_init<<<256, 128, 0, stream>>>(h0, c0, h0s, c0s, c1s, hb[0], ac[0]);
  k_emb<<<8192, 256, 0, stream>>>(emb, tw, EmbX);
  k_cvt<<<16384, 256, 0, stream>>>(Whh0, 1024, Whh0bf, 1024, 4096, 1024);
  k_cvt<<<16384, 256, 0, stream>>>(Wih1, 1024, W1cat, 2048, 4096, 1024);
  k_cvt<<<16384, 256, 0, stream>>>(Whh1, 1024, W1cat + 1024, 2048, 4096, 1024);

  // Hoisted input projection for layer 0: [emb|context] @ Wih0^T + (bih0+bhh0)
  gemm_bf16_wmma<true, true, false><<<dim3(64, 32), 128, 0, stream>>>(
      EmbX, 1024, ctx_in, 1024, 1024, Wih0, 2048, b0c, Xproj, 4096, 2048);

  // Recurrence: 64 steps x 2 layers, h ping-pong, fused gates+pointwise
  for (int t = 0; t < 64; ++t) {
    int cur = t & 1, nxt = cur ^ 1;
    lstm_step<true, false, true><<<64, 128, 0, stream>>>(
        hb[cur], 1024, Whh0bf, Xproj + (size_t)t * 32 * 4096, nullptr,
        c0s, h0s, hb[nxt], 1024, ac[cur], 2048);
    lstm_step<false, true, false><<<64, 128, 0, stream>>>(
        ac[cur], 2048, W1cat, nullptr, b1c,
        c1s, y1 + (size_t)t * 32 * 1024, ac[nxt] + 1024, 2048, nullptr, 0);
  }

  // Attention
  gemm_bf16_wmma<false, true, false><<<dim3(8, 32), 128, 0, stream>>>(
      y1, 1024, nullptr, 0, 1024, aw1, 1024, ab1, attnp, 512, 1024);
  k_pred<<<2048, 128, 0, stream>>>(attnp, aw2, ab2, lens, p_arr, ws_arr);
  k_attn<<<2048, 128, 0, stream>>>(enc, y1, p_arr, ws_arr, lens, ctx_out);

  // Output head: fc1 (relu, split-A concat [ctx|y1]) then fc2 vocab projection
  gemm_bf16_wmma<true, true, true><<<dim3(16, 32), 128, 0, stream>>>(
      ctx_out, 1024, y1, 1024, 1024, fc1w, 2048, fc1b, fc1o, 1024, 2048);
  gemm_bf16_wmma<false, true, false><<<dim3(500, 32), 128, 0, stream>>>(
      fc1o, 1024, nullptr, 0, 1024, fc2w, 1024, fc2b, dout, 32000, 1024);

  // Final h_n / c_n
  k_fin<<<256, 128, 0, stream>>>(h0s, y1 + (size_t)63 * 32768, c0s, c1s,
                                 dout + OFF_HN, dout + OFF_CN);

  (void)in_sizes; (void)n_in; (void)out_size; (void)ws_size;
}
